// Contraction_35777077576017
// MI455X (gfx1250) — compile-verified, matrix operand order checked by reading, not run on hardware
//
#include <hip/hip_runtime.h>
#include <math.h>

// ---------------------------------------------------------------------------
// Fused equivariant tensor-product layer for MI455X (gfx1250).
//  - per-edge MLP + 23 channel-wise TP contractions in f32 VALU
//  - cat buffers in LDS as bf16 (258KB; CDNA5 has 320KB/WGP)
//  - final (E*3^ro, Cin) x (Cin, 64) linears via v_wmma_f32_16x16x32_bf16
// Roofline: ~620MB mandatory traffic (~27us @ 23.3TB/s); the 40GFLOP linear
// rides the WMMA pipe (f32 VALU would be ~150us), bf16 inputs / f32 accum.
// ---------------------------------------------------------------------------

typedef __attribute__((ext_vector_type(16))) __bf16 v16bf;
typedef __attribute__((ext_vector_type(8)))  __bf16 v8bf;
typedef __attribute__((ext_vector_type(8)))  float  v8f;

constexpr int E_TOT   = 20000;
constexpr int TILE_E  = 8;                 // edges per workgroup
constexpr int NBLK    = E_TOT / TILE_E;    // 2500

constexpr int ipow3(int n) { return n <= 0 ? 1 : 3 * ipow3(n - 1); }

// LDS layout (bytes), all 16B aligned:
//   hsm   @      0 : 8*64 f32        (2048)
//   convw @   2048 : 8*1472 f32      (47104)
//   cat0  @  49152 : 8  rows * 264 bf16  (4224)   Cin=256 (+8 pad)
//   cat1  @  53376 : 24 rows * 392 bf16  (18816)  Cin=384
//   cat2  @  72192 : 72 rows * 456 bf16  (65664)  Cin=448
//   cat3  @ 137856 : 216rows * 392 bf16  (169344) Cin=384
// total 307200 B
constexpr unsigned SMEM_BYTES = 307200;

template<int R>
__device__ __forceinline__ void load_vec(const float* __restrict__ p, long ec, float* v) {
  constexpr int T = ipow3(R);
  const float* q = p + ec * T;
  #pragma unroll
  for (int t = 0; t < T; ++t) v[t] = q[t];
}

// out[a*P2+b] = sum_s x[a*K3+s] * y[b*K3+s]   (contract last k axes of each)
template<int R1, int R2, int RO, int G>
__device__ __forceinline__ void tp_combo(const float* xr, const float* yr, float w,
                                         __bf16* __restrict__ cat, int eloc, int c) {
  constexpr int K  = (R1 + R2 - RO) / 2;
  constexpr int K3 = ipow3(K);
  constexpr int P1 = ipow3(R1 - K);
  constexpr int P2 = ipow3(R2 - K);
  constexpr int RS = (RO == 0) ? 264 : (RO == 1) ? 392 : (RO == 2) ? 456 : 392;
  constexpr int T3 = ipow3(RO);
  #pragma unroll
  for (int a = 0; a < P1; ++a) {
    #pragma unroll
    for (int b = 0; b < P2; ++b) {
      float acc = 0.f;
      #pragma unroll
      for (int s = 0; s < K3; ++s) acc += xr[a * K3 + s] * yr[b * K3 + s];
      cat[(eloc * T3 + a * P2 + b) * RS + G * 64 + c] = (__bf16)(acc * w);
    }
  }
}

__global__ __launch_bounds__(256, 1) void fused_tp_kernel(
    const float* __restrict__ x0, const float* __restrict__ x1,
    const float* __restrict__ x2, const float* __restrict__ x3,
    const float* __restrict__ y0, const float* __restrict__ y1,
    const float* __restrict__ y2, const float* __restrict__ y3,
    const float* __restrict__ radial, const float* __restrict__ cutoff,
    const float* __restrict__ W1, const float* __restrict__ W2,
    const __bf16* __restrict__ ldT,   // bf16 transposed ld weights in d_ws
    float* __restrict__ out)
{
  extern __shared__ char smem_raw[];
  float*  hsm   = reinterpret_cast<float*>(smem_raw);
  float*  convw = reinterpret_cast<float*>(smem_raw + 2048);
  __bf16* cat0  = reinterpret_cast<__bf16*>(smem_raw + 49152);
  __bf16* cat1  = reinterpret_cast<__bf16*>(smem_raw + 53376);
  __bf16* cat2  = reinterpret_cast<__bf16*>(smem_raw + 72192);
  __bf16* cat3  = reinterpret_cast<__bf16*>(smem_raw + 137856);

  const int tid = threadIdx.x;
  const int e0  = blockIdx.x * TILE_E;

  // ---- Phase 1: h = silu(radial @ W1), 8 edges x 64 ----
  #pragma unroll
  for (int q = 0; q < 2; ++q) {
    int t = q * 256 + tid;
    int eloc = t >> 6, j = t & 63;
    float s = 0.f;
    #pragma unroll
    for (int n = 0; n < 8; ++n) s += radial[(e0 + eloc) * 8 + n] * W1[n * 64 + j];
    hsm[t] = s / (1.f + __expf(-s));
  }
  __syncthreads();

  // ---- Phase 2: convw = (h @ W2) * cutoff, 8 edges x 1472 (W2 is L2-resident) ----
  for (int eloc = 0; eloc < 8; ++eloc) {
    float cut = cutoff[e0 + eloc];
    for (int j = tid; j < 1472; j += 256) {
      float acc = 0.f;
      #pragma unroll
      for (int kk = 0; kk < 64; ++kk) acc += hsm[eloc * 64 + kk] * W2[kk * 1472 + j];
      convw[eloc * 1472 + j] = acc * cut;
    }
  }
  __syncthreads();

  // ---- Phase 3: 23 tensor-product contractions -> weighted bf16 cat in LDS ----
  #pragma unroll 1
  for (int q = 0; q < 2; ++q) {
    int ecl  = q * 256 + tid;              // 512 (edge, channel) pairs
    int eloc = ecl >> 6, c = ecl & 63;
    long ec  = (long)(e0 + eloc) * 64 + c;
    const float* cw = convw + eloc * 1472 + c;
    float xr[27], yr[27];
    // r1 = 0
    load_vec<0>(x0, ec, xr);
    load_vec<0>(y0, ec, yr); tp_combo<0,0,0,0>(xr, yr, cw[ 0*64], cat0, eloc, c);
    load_vec<1>(y1, ec, yr); tp_combo<0,1,1,0>(xr, yr, cw[ 1*64], cat1, eloc, c);
    load_vec<2>(y2, ec, yr); tp_combo<0,2,2,0>(xr, yr, cw[ 2*64], cat2, eloc, c);
    load_vec<3>(y3, ec, yr); tp_combo<0,3,3,0>(xr, yr, cw[ 3*64], cat3, eloc, c);
    // r1 = 1
    load_vec<1>(x1, ec, xr);
    load_vec<0>(y0, ec, yr); tp_combo<1,0,1,1>(xr, yr, cw[ 4*64], cat1, eloc, c);
    load_vec<1>(y1, ec, yr); tp_combo<1,1,0,1>(xr, yr, cw[ 5*64], cat0, eloc, c);
                             tp_combo<1,1,2,1>(xr, yr, cw[ 6*64], cat2, eloc, c);
    load_vec<2>(y2, ec, yr); tp_combo<1,2,1,2>(xr, yr, cw[ 7*64], cat1, eloc, c);
                             tp_combo<1,2,3,1>(xr, yr, cw[ 8*64], cat3, eloc, c);
    load_vec<3>(y3, ec, yr); tp_combo<1,3,2,2>(xr, yr, cw[ 9*64], cat2, eloc, c);
    // r1 = 2
    load_vec<2>(x2, ec, xr);
    load_vec<0>(y0, ec, yr); tp_combo<2,0,2,3>(xr, yr, cw[10*64], cat2, eloc, c);
    load_vec<1>(y1, ec, yr); tp_combo<2,1,1,3>(xr, yr, cw[11*64], cat1, eloc, c);
                             tp_combo<2,1,3,2>(xr, yr, cw[12*64], cat3, eloc, c);
    load_vec<2>(y2, ec, yr); tp_combo<2,2,0,2>(xr, yr, cw[13*64], cat0, eloc, c);
                             tp_combo<2,2,2,4>(xr, yr, cw[14*64], cat2, eloc, c);
    load_vec<3>(y3, ec, yr); tp_combo<2,3,1,4>(xr, yr, cw[15*64], cat1, eloc, c);
                             tp_combo<2,3,3,3>(xr, yr, cw[16*64], cat3, eloc, c);
    // r1 = 3
    load_vec<3>(x3, ec, xr);
    load_vec<0>(y0, ec, yr); tp_combo<3,0,3,4>(xr, yr, cw[17*64], cat3, eloc, c);
    load_vec<1>(y1, ec, yr); tp_combo<3,1,2,5>(xr, yr, cw[18*64], cat2, eloc, c);
    load_vec<2>(y2, ec, yr); tp_combo<3,2,1,5>(xr, yr, cw[19*64], cat1, eloc, c);
                             tp_combo<3,2,3,5>(xr, yr, cw[20*64], cat3, eloc, c);
    load_vec<3>(y3, ec, yr); tp_combo<3,3,0,3>(xr, yr, cw[21*64], cat0, eloc, c);
                             tp_combo<3,3,2,6>(xr, yr, cw[22*64], cat2, eloc, c);
  }
  __syncthreads();

  // ---- Phase 4: per-ro linears via WMMA bf16 (f32 accumulate) ----
  // GEMM_ro: M = 8*3^ro rows (edge,spat), N = 64, K = Cin(ro).
  const int wave = tid >> 5;
  const int lane = tid & 31;
  const int half = lane >> 4;       // which K-half this lane owns
  const int l15  = lane & 15;

  // 88 (mtile,ntile) jobs total: 4(ro0)+8(ro1)+20(ro2)+56(ro3).
  // job is wave-uniform by construction -> force it scalar so all per-job
  // constants live in SGPRs and the K loop uses scalar control flow.
  for (int jobv = wave; jobv < 88; jobv += 8) {
    const int job = __builtin_amdgcn_readfirstlane(jobv);
    // Decode job -> per-ro constants as scalar selects (no rodata table loads).
    int local, kts, rows, cin, rs, p3, magic, ldoff;
    long obase;
    const __bf16* catp;
    if (job < 4) {
      local = job;      kts = 8;  rows = 8;   cin = 256; rs = 264; p3 = 1;
      magic = 512; ldoff = 0;     obase = 0L;                     catp = cat0;
    } else if (job < 12) {
      local = job - 4;  kts = 12; rows = 24;  cin = 384; rs = 392; p3 = 3;
      magic = 171; ldoff = 16384; obase = (long)E_TOT * 64;       catp = cat1;
    } else if (job < 32) {
      local = job - 12; kts = 14; rows = 72;  cin = 448; rs = 456; p3 = 9;
      magic = 57;  ldoff = 40960; obase = (long)E_TOT * 64 * 4;   catp = cat2;
    } else {
      local = job - 32; kts = 12; rows = 216; cin = 384; rs = 392; p3 = 27;
      magic = 19;  ldoff = 69632; obase = (long)E_TOT * 64 * 13;  catp = cat3;
    }
    const int mtile = local >> 2, ntile = local & 3;
    const __bf16* ldw = ldT + ldoff;

    int arow = mtile * 16 + l15;
    if (arow >= rows) arow = 0;                 // clamp padded rows (stores predicated)
    const __bf16* abase = catp + arow * rs + half * 8;                   // LDS
    const __bf16* bbase = ldw + (ntile * 16 + l15) * cin + half * 8;     // global (L2)

    // Plain K loop: the 4 fragment loads of iteration i+1 issue right after
    // the WMMA of iteration i consumes its operands, overlapping naturally.
    v8f acc = {0.f, 0.f, 0.f, 0.f, 0.f, 0.f, 0.f, 0.f};
    for (int kt = 0; kt < kts; ++kt) {
      v8bf alo = *reinterpret_cast<const v8bf*>(abase + kt * 32);
      v8bf ahi = *reinterpret_cast<const v8bf*>(abase + kt * 32 + 16);
      v8bf blo = *reinterpret_cast<const v8bf*>(bbase + kt * 32);
      v8bf bhi = *reinterpret_cast<const v8bf*>(bbase + kt * 32 + 16);
      v16bf a, b;
      #pragma unroll
      for (int i = 0; i < 8; ++i) {
        a[i] = alo[i]; a[i + 8] = ahi[i];
        b[i] = blo[i]; b[i + 8] = bhi[i];
      }
      acc = __builtin_amdgcn_wmma_f32_16x16x32_bf16(
          /*neg_a=*/false, a, /*neg_b=*/false, b,
          /*c_mod=*/(short)0, acc, /*reuse_a=*/false, /*reuse_b=*/false);
    }

    // Store D: element (m,n) -> out[obase + ((e0+eloc)*64 + hcol)*p3 + spat].
    // 64-bit part hoisted per job; per-element offset is small 32-bit math.
    const int  hcol = ntile * 16 + l15;         // output channel
    const int  p364 = p3 * 64;
    float* __restrict__ optr = out + (obase + ((long)e0 * 64 + hcol) * p3);
    #pragma unroll
    for (int v = 0; v < 8; ++v) {
      int grow = mtile * 16 + v + 8 * half;     // D layout: M = v + 8*(lane>>4)
      if (grow < rows) {
        int eloc = (grow * magic) >> 9;         // exact /p3 for p3 in {1,3,9,27}
        int spat = grow - eloc * p3;
        optr[eloc * p364 + spat] = acc[v];
      }
    }
  }
}

// Convert ld0..ld3 (Cin,64) f32 -> ldT (64,Cin) bf16 in workspace (B-operand layout).
__global__ void cvt_ld_kernel(const float* __restrict__ ld0, const float* __restrict__ ld1,
                              const float* __restrict__ ld2, const float* __restrict__ ld3,
                              __bf16* __restrict__ ws) {
  int idx = blockIdx.x * blockDim.x + threadIdx.x;
  const float* src; __bf16* dst; int cin;
  if (idx < 16384)      { src = ld0; dst = ws;          cin = 256; }
  else if (idx < 40960) { src = ld1; dst = ws + 16384;  cin = 384; idx -= 16384; }
  else if (idx < 69632) { src = ld2; dst = ws + 40960;  cin = 448; idx -= 40960; }
  else if (idx < 94208) { src = ld3; dst = ws + 69632;  cin = 384; idx -= 69632; }
  else return;
  int h = idx / cin, c = idx - h * cin;
  dst[h * cin + c] = (__bf16)src[c * 64 + h];
}

extern "C" void kernel_launch(void* const* d_in, const int* in_sizes, int n_in,
                              void* d_out, int out_size, void* d_ws, size_t ws_size,
                              hipStream_t stream) {
  (void)in_sizes; (void)n_in; (void)out_size; (void)ws_size;
  const float* x0 = (const float*)d_in[0];
  const float* x1 = (const float*)d_in[1];
  const float* x2 = (const float*)d_in[2];
  const float* x3 = (const float*)d_in[3];
  const float* y0 = (const float*)d_in[4];
  const float* y1 = (const float*)d_in[5];
  const float* y2 = (const float*)d_in[6];
  const float* y3 = (const float*)d_in[7];
  const float* radial = (const float*)d_in[8];
  const float* cutoff = (const float*)d_in[9];
  const float* W1  = (const float*)d_in[10];
  const float* W2  = (const float*)d_in[11];
  const float* ld0 = (const float*)d_in[12];
  const float* ld1 = (const float*)d_in[13];
  const float* ld2 = (const float*)d_in[14];
  const float* ld3 = (const float*)d_in[15];
  __bf16* ldT = (__bf16*)d_ws;   // needs 94208 * 2 = 188,416 bytes

  cvt_ld_kernel<<<(94208 + 255) / 256, 256, 0, stream>>>(ld0, ld1, ld2, ld3, ldT);
  fused_tp_kernel<<<NBLK, 256, SMEM_BYTES, stream>>>(
      x0, x1, x2, x3, y0, y1, y2, y3, radial, cutoff, W1, W2, ldT, (float*)d_out);
}
